// AttentionLayer_51642686767196
// MI455X (gfx1250) — matrix operand order repaired and checked by reading, not compiled, not run on hardware
//
#include <hip/hip_runtime.h>
#include <hip/hip_bf16.h>

#define BDIM 4
#define SDIM 2048
#define DDIM 1024
#define HNUM 16
#define DH   64

typedef __attribute__((ext_vector_type(16))) __bf16          v16bf;
typedef __attribute__((ext_vector_type(8)))  float           v8f;
typedef __attribute__((ext_vector_type(8)))  unsigned short  v8us;
typedef __attribute__((ext_vector_type(16))) unsigned short  v16us;
typedef __attribute__((ext_vector_type(4)))  int             v4i;
typedef __attribute__((address_space(1)))    v4i             g_v4i;   // global int4
typedef __attribute__((address_space(3)))    v4i             l_v4i;   // LDS int4

#if defined(__has_builtin)
#  if __has_builtin(__builtin_amdgcn_global_load_async_to_lds_b128)
#    define HAS_ASYNC_LDS 1
#  endif
#endif

__device__ __forceinline__ unsigned short f32_to_bf16(float f) {
  unsigned int u = __float_as_uint(f);
  unsigned int r = (u + 0x7fffu + ((u >> 16) & 1u)) >> 16;   // round-to-nearest-even
  return (unsigned short)r;
}

__device__ __forceinline__ v16bf make_frag(v8us lo, v8us hi) {
  v16us c = __builtin_shufflevector(lo, hi, 0,1,2,3,4,5,6,7,8,9,10,11,12,13,14,15);
  return __builtin_bit_cast(v16bf, c);
}

// A-matrix 16x32 bf16 fragment. ISA layout: lanes 0-15 hold K {0..7,16..23}; lanes 16-31 hold K {8..15,24..31}.
__device__ __forceinline__ v16bf load_a_frag(const unsigned short* base, int ld, int row0, int kb, int lane) {
  int m  = row0 + (lane & 15);
  int ko = kb + ((lane & 16) ? 8 : 0);
  const unsigned short* p = base + (size_t)m * ld + ko;
  return make_frag(*(const v8us*)p, *(const v8us*)(p + 16));
}

// B-matrix 32x16 bf16 fragment where B[k][n] = srcT[n][k]. Lanes 0-15: col n, K {0..7,16..23}; lanes 16-31: col n, K {8..15,24..31}.
__device__ __forceinline__ v16bf load_b_frag(const unsigned short* baseT, int ld, int col0, int kb, int lane) {
  int n  = col0 + (lane & 15);
  int ko = kb + ((lane & 16) ? 8 : 0);
  const unsigned short* p = baseT + (size_t)n * ld + ko;
  return make_frag(*(const v8us*)p, *(const v8us*)(p + 16));
}

__device__ __forceinline__ v8f wmma_bf16(v16bf a, v16bf b, v8f c) {
  return __builtin_amdgcn_wmma_f32_16x16x32_bf16(false, a, false, b, (short)0, c, false, false);
}

// ---- async global->LDS staging (gfx1250 GLOBAL_LOAD_ASYNC_TO_LDS_B128, ASYNCcnt) ----
__device__ __forceinline__ void stage16(unsigned short* lds_dst, const unsigned short* gsrc) {
#if defined(HAS_ASYNC_LDS)
  unsigned short* gs = const_cast<unsigned short*>(gsrc);
  __builtin_amdgcn_global_load_async_to_lds_b128((g_v4i*)gs, (l_v4i*)lds_dst, 0, 0);
#else
  *(v8us*)lds_dst = *(const v8us*)gsrc;     // synchronous fallback
#endif
}
__device__ __forceinline__ void async_wait_le2() {
#if defined(HAS_ASYNC_LDS)
#  if __has_builtin(__builtin_amdgcn_s_wait_asynccnt)
  __builtin_amdgcn_s_wait_asynccnt(2);
#  else
  asm volatile("s_wait_asynccnt 0x2" ::: "memory");
#  endif
#endif
}
__device__ __forceinline__ void async_wait_0() {
#if defined(HAS_ASYNC_LDS)
#  if __has_builtin(__builtin_amdgcn_s_wait_asynccnt)
  __builtin_amdgcn_s_wait_asynccnt(0);
#  else
  asm volatile("s_wait_asynccnt 0x0" ::: "memory");
#  endif
#endif
}

// ---------------- f32 -> bf16 converts ----------------
__global__ void cvt_kernel(const float* __restrict__ in, unsigned short* __restrict__ outp, int n) {
  int i = blockIdx.x * blockDim.x + threadIdx.x;
  if (i < n) outp[i] = f32_to_bf16(in[i]);
}

// weights [K=1024][N=1024] f32 -> [N][K] bf16
__global__ void cvt_t_kernel(const float* __restrict__ in, unsigned short* __restrict__ outp) {
  int i = blockIdx.x * blockDim.x + threadIdx.x;   // i < 1024*1024
  int k = i >> 10, n = i & (DDIM - 1);
  outp[(size_t)n * DDIM + k] = f32_to_bf16(in[i]);
}

// ---------------- QKV projection GEMM ----------------
// C[8192,1024] = A[8192,1024] x W[1024,1024] + bias. mode: 0=Q(*0.125), 1=K, 2=V(store transposed).
__global__ __launch_bounds__(256) void gemm_qkv_kernel(
    const unsigned short* __restrict__ A,
    const unsigned short* __restrict__ Wt,
    const float* __restrict__ bias,
    unsigned short* __restrict__ outp,
    int mode)
{
  const int lane = threadIdx.x & 31;
  const int wave = threadIdx.x >> 5;
  const int m0 = blockIdx.x * 128 + (wave >> 1) * 32;
  const int n0 = blockIdx.y * 128 + (wave & 1) * 64;
  const int nl = lane & 15;
  const int hb = (lane & 16) ? 8 : 0;

  v8f acc[2][4];
#pragma unroll
  for (int i = 0; i < 2; i++)
#pragma unroll
    for (int j = 0; j < 4; j++)
      acc[i][j] = (v8f){0.f, 0.f, 0.f, 0.f, 0.f, 0.f, 0.f, 0.f};

  for (int kb = 0; kb < DDIM; kb += 32) {
    v16bf a0 = load_a_frag(A, DDIM, m0,      kb, lane);
    v16bf a1 = load_a_frag(A, DDIM, m0 + 16, kb, lane);
#pragma unroll
    for (int j = 0; j < 4; j++) {
      v16bf bf = load_b_frag(Wt, DDIM, n0 + j * 16, kb, lane);
      acc[0][j] = wmma_bf16(a0, bf, acc[0][j]);
      acc[1][j] = wmma_bf16(a1, bf, acc[1][j]);
    }
  }

#pragma unroll
  for (int i = 0; i < 2; i++)
#pragma unroll
    for (int j = 0; j < 4; j++)
#pragma unroll
      for (int r = 0; r < 8; r++) {
        const int m = m0 + i * 16 + r + hb;
        const int n = n0 + j * 16 + nl;
        float v = acc[i][j][r] + bias[n];
        if (mode == 0) v *= 0.125f;                 // fold 1/sqrt(DH) into Q (exact in bf16)
        const int bb = m >> 11, s = m & (SDIM - 1);
        const int hh = n >> 6,  d = n & (DH - 1);
        const unsigned short bv = f32_to_bf16(v);
        if (mode == 2)
          outp[(((size_t)bb * HNUM + hh) * DH + d) * SDIM + s] = bv;     // V transposed [b,h,dh,s]
        else
          outp[(((size_t)bb * HNUM + hh) * SDIM + s) * DH + d] = bv;     // Q/K [b,h,s,dh]
      }
}

// ---------------- flash attention ----------------
// Grid: (S/128, B*H). 8 waves/block; wave owns a 16-row f strip. K/V tiles double-buffered
// in LDS via async global->LDS copies shared by the whole workgroup (8x traffic reduction).
#define KLD 72   // 64 + 8 pad: 36-dword row stride -> 16 lanes hit 64 distinct banks
#define VLD 40   // 32 + 8 pad: 20-dword row stride -> conflict-free
#define PLD 40

__global__ __launch_bounds__(256) void attn_kernel(
    const unsigned short* __restrict__ Q,    // [b,h,s,dh] bf16 (pre-scaled by 0.125)
    const unsigned short* __restrict__ K,    // [b,h,s,dh] bf16
    const unsigned short* __restrict__ Vt,   // [b,h,dh,s] bf16
    const int* __restrict__ mask,            // [b,s,s]
    float* __restrict__ out)                 // [b,s,h*dh]
{
  __shared__ unsigned short pbuf[8][16 * PLD];
  __shared__ unsigned short kbuf[2][32 * KLD];
  __shared__ unsigned short vbuf[2][64 * VLD];

  const int tid  = threadIdx.x;
  const int lane = tid & 31;
  const int wave = tid >> 5;
  const int nl   = lane & 15;
  const int hb   = (lane & 16) ? 8 : 0;
  const int bh   = blockIdx.y;
  const int b    = bh >> 4;
  const int h    = bh & 15;
  const int f0   = blockIdx.x * 128 + wave * 16;

  const unsigned short* Qp = Q  + (size_t)bh * SDIM * DH;
  const unsigned short* Kp = K  + (size_t)bh * SDIM * DH;
  const unsigned short* Vp = Vt + (size_t)bh * DH * SDIM;
  const int* Mp = mask + (size_t)b * SDIM * SDIM;

  // cooperative staging coordinates: K tile 32x64 (8 thr/row), V tile 64x32 (4 thr/row)
  const int krow = tid >> 3, kcol = (tid & 7) * 8;
  const int vrow = tid >> 2, vcol = (tid & 3) * 8;

  const v16bf qa0 = load_a_frag(Qp, DH, f0, 0,  lane);
  const v16bf qa1 = load_a_frag(Qp, DH, f0, 32, lane);

  v8f o0 = {0.f,0.f,0.f,0.f,0.f,0.f,0.f,0.f}, o1 = o0, o2 = o0, o3 = o0;
  float mrow[8], lrow[8];
#pragma unroll
  for (int r = 0; r < 8; r++) { mrow[r] = -1e30f; lrow[r] = 0.f; }

  unsigned short* myp = pbuf[wave];

  // prologue: stage chunk 0 into buffer 0
  stage16(&kbuf[0][krow * KLD + kcol], Kp + (size_t)krow * DH + kcol);
  stage16(&vbuf[0][vrow * VLD + vcol], Vp + (size_t)vrow * SDIM + vcol);

  for (int t0 = 0; t0 < SDIM; t0 += 32) {
    const int db = (t0 >> 5) & 1;
    if (t0 + 32 < SDIM) {   // stage next chunk into other buffer
      const int tn = t0 + 32;
      stage16(&kbuf[db ^ 1][krow * KLD + kcol], Kp + (size_t)(tn + krow) * DH + kcol);
      stage16(&vbuf[db ^ 1][vrow * VLD + vcol], Vp + (size_t)vrow * SDIM + tn + vcol);
      async_wait_le2();     // chunk t0's copies complete (in-order), next chunk in flight
    } else {
      async_wait_0();
    }
    __syncthreads();        // all waves' slices of chunk t0 visible

    v8f s0 = {0.f,0.f,0.f,0.f,0.f,0.f,0.f,0.f};
    v8f s1 = s0;
    {
      v16bf k0 = load_b_frag(kbuf[db], KLD, 0, 0,  lane);
      v16bf k1 = load_b_frag(kbuf[db], KLD, 0, 32, lane);
      s0 = wmma_bf16(qa0, k0, s0);
      s0 = wmma_bf16(qa1, k1, s0);
    }
    {
      v16bf k0 = load_b_frag(kbuf[db], KLD, 16, 0,  lane);
      v16bf k1 = load_b_frag(kbuf[db], KLD, 16, 32, lane);
      s1 = wmma_bf16(qa0, k0, s1);
      s1 = wmma_bf16(qa1, k1, s1);
    }

#pragma unroll
    for (int r = 0; r < 8; r++) {
      const int f = f0 + r + hb;
      const int* mp = Mp + (size_t)f * SDIM + t0;
      float sv0 = s0[r] + (1.0f - (float)mp[nl])      * -10000.0f;
      float sv1 = s1[r] + (1.0f - (float)mp[nl + 16]) * -10000.0f;
      float mx = fmaxf(sv0, sv1);
      mx = fmaxf(mx, __shfl_xor(mx, 1, 16));
      mx = fmaxf(mx, __shfl_xor(mx, 2, 16));
      mx = fmaxf(mx, __shfl_xor(mx, 4, 16));
      mx = fmaxf(mx, __shfl_xor(mx, 8, 16));
      const float mnew  = fmaxf(mrow[r], mx);
      const float alpha = __expf(mrow[r] - mnew);
      const float p0 = __expf(sv0 - mnew);
      const float p1 = __expf(sv1 - mnew);
      float rs = p0 + p1;
      rs += __shfl_xor(rs, 1, 16);
      rs += __shfl_xor(rs, 2, 16);
      rs += __shfl_xor(rs, 4, 16);
      rs += __shfl_xor(rs, 8, 16);
      lrow[r] = lrow[r] * alpha + rs;
      mrow[r] = mnew;
      o0[r] *= alpha; o1[r] *= alpha; o2[r] *= alpha; o3[r] *= alpha;
      myp[(r + hb) * PLD + nl]      = f32_to_bf16(p0);
      myp[(r + hb) * PLD + nl + 16] = f32_to_bf16(p1);
    }
    asm volatile("" ::: "memory");          // order wave-private P stores before fragment reload
    const v16bf pa = load_a_frag(myp, PLD, 0, 0, lane);
    o0 = wmma_bf16(pa, load_b_frag(vbuf[db], VLD, 0,  0, lane), o0);
    o1 = wmma_bf16(pa, load_b_frag(vbuf[db], VLD, 16, 0, lane), o1);
    o2 = wmma_bf16(pa, load_b_frag(vbuf[db], VLD, 32, 0, lane), o2);
    o3 = wmma_bf16(pa, load_b_frag(vbuf[db], VLD, 48, 0, lane), o3);

    __syncthreads();        // everyone done reading chunk t0 before its buffer is restaged
  }

#pragma unroll
  for (int r = 0; r < 8; r++) {
    const int f = f0 + r + hb;
    const float inv = 1.0f / lrow[r];
    float* op = out + ((size_t)b * SDIM + f) * DDIM + h * DH + nl;
    op[0]  = o0[r] * inv;
    op[16] = o1[r] * inv;
    op[32] = o2[r] * inv;
    op[48] = o3[r] * inv;
  }
}

extern "C" void kernel_launch(void* const* d_in, const int* in_sizes, int n_in,
                              void* d_out, int out_size, void* d_ws, size_t ws_size,
                              hipStream_t stream) {
  const float* x_from = (const float*)d_in[0];
  const float* x_to   = (const float*)d_in[1];
  const int*   amask  = (const int*)d_in[2];
  const float* wq = (const float*)d_in[3];
  const float* bq = (const float*)d_in[4];
  const float* wk = (const float*)d_in[5];
  const float* bk = (const float*)d_in[6];
  const float* wv = (const float*)d_in[7];
  const float* bv = (const float*)d_in[8];
  float* out = (float*)d_out;

  const size_t NX = (size_t)BDIM * SDIM * DDIM;   // 8,388,608 elements
  const size_t NW = (size_t)DDIM * DDIM;          // 1,048,576 elements
  unsigned short* ws  = (unsigned short*)d_ws;    // ~86 MB bf16 scratch
  unsigned short* xf  = ws;
  unsigned short* xt  = xf  + NX;
  unsigned short* wtq = xt  + NX;
  unsigned short* wtk = wtq + NW;
  unsigned short* wtv = wtk + NW;
  unsigned short* Qb  = wtv + NW;
  unsigned short* Kb  = Qb  + NX;
  unsigned short* Vb  = Kb  + NX;

  cvt_kernel<<<(int)((NX + 255) / 256), 256, 0, stream>>>(x_from, xf, (int)NX);
  cvt_kernel<<<(int)((NX + 255) / 256), 256, 0, stream>>>(x_to,   xt, (int)NX);
  cvt_t_kernel<<<(int)(NW / 256), 256, 0, stream>>>(wq, wtq);
  cvt_t_kernel<<<(int)(NW / 256), 256, 0, stream>>>(wk, wtk);
  cvt_t_kernel<<<(int)(NW / 256), 256, 0, stream>>>(wv, wtv);

  dim3 gg(BDIM * SDIM / 128, DDIM / 128);   // (64, 8)
  gemm_qkv_kernel<<<gg, 256, 0, stream>>>(xf, wtq, bq, Qb, 0);
  gemm_qkv_kernel<<<gg, 256, 0, stream>>>(xt, wtk, bk, Kb, 1);
  gemm_qkv_kernel<<<gg, 256, 0, stream>>>(xt, wtv, bv, Vb, 2);

  dim3 ga(SDIM / 128, BDIM * HNUM);         // (16, 64)
  attn_kernel<<<ga, 256, 0, stream>>>(Qb, Kb, Vb, amask, out);
}
